// lstm_double_17489106829722
// MI455X (gfx1250) — matrix operand order repaired and checked by reading, not compiled, not run on hardware
//
#include <hip/hip_runtime.h>

typedef _Float16 f16;
typedef __attribute__((ext_vector_type(16))) _Float16 v16h;
typedef __attribute__((ext_vector_type(8)))  _Float16 v8h;
typedef __attribute__((ext_vector_type(4)))  _Float16 v4h;
typedef __attribute__((ext_vector_type(8)))  float    v8f;
typedef __attribute__((ext_vector_type(4)))  float    v4f;
typedef __attribute__((ext_vector_type(4)))  unsigned int u32x4;
typedef __attribute__((ext_vector_type(8)))  int i32x8;
typedef __attribute__((ext_vector_type(4)))  int i32x4;

#define BATCH 32
#define TLEN  2048
#define HID   256
#define GATES 1024   // 4*HID
#define NWG   8      // workgroups per direction
#define HS    32     // hidden units per WG
#define LCOLS 128    // local gate columns per WG (32 each of i,f,g,o)
#define GP    132    // padded gate-buffer stride (banks)

#if __has_builtin(__builtin_amdgcn_tensor_load_to_lds)
#define USE_TDM 1
#else
#define USE_TDM 0
#endif

// ---------------------------------------------------------------------------
// Pack W = [W_ih | W_hh] (f32, row = gate) into per-lane WMMA B-fragment
// layout, f16:  PW[wg][kb][ct][lane][16]
// ---------------------------------------------------------------------------
__global__ void pack_weights(const float* __restrict__ w_ih,
                             const float* __restrict__ w_hh,
                             int Din, int K, f16* __restrict__ PW) {
    int f = blockIdx.x * blockDim.x + threadIdx.x;
    if (f >= K * GATES) return;
    int K32  = K >> 5;
    int hidx = f & 15;
    int t1   = f >> 4;
    int lane = t1 & 31;
    int t2   = t1 >> 5;
    int ct   = t2 & 7;
    int t3   = t2 >> 3;
    int kb   = t3 % K32;
    int wg   = t3 / K32;
    int n    = lane & 15;
    int lc   = ct * 16 + n;
    int gt   = lc >> 5;
    int j    = lc & 31;
    int row  = gt * HID + wg * HS + j;
    int k    = kb * 32 + (hidx & 8) * 2 + (lane >> 4) * 8 + (hidx & 7);
    float v  = (k < Din) ? w_ih[(size_t)row * Din + k]
                         : w_hh[(size_t)row * HID + (k - Din)];
    PW[f] = (f16)v;
}

__global__ void bias_sum(const float* __restrict__ bi,
                         const float* __restrict__ bh,
                         float* __restrict__ out) {
    int g = blockIdx.x * blockDim.x + threadIdx.x;
    if (g < GATES) out[g] = bi[g] + bh[g];
}

// x (B,T,128) f32 -> f16 (TDM cannot convert)
__global__ void cvt_f32_to_f16(const float* __restrict__ src, f16* __restrict__ dst, int n4) {
    int i = blockIdx.x * blockDim.x + threadIdx.x;
    if (i >= n4) return;
    v4f v = ((const v4f*)src)[i];
    v4h h;
    h[0] = (f16)v[0]; h[1] = (f16)v[1]; h[2] = (f16)v[2]; h[3] = (f16)v[3];
    ((v4h*)dst)[i] = h;
}

__device__ __forceinline__ void grid_barrier(int* cnt, int target) {
    __threadfence();
    __syncthreads();
    if (threadIdx.x == 0) {
        __hip_atomic_fetch_add(cnt, 1, __ATOMIC_RELEASE, __HIP_MEMORY_SCOPE_AGENT);
        while (__hip_atomic_load(cnt, __ATOMIC_ACQUIRE, __HIP_MEMORY_SCOPE_AGENT) < target)
            __builtin_amdgcn_s_sleep(2);
    }
    __syncthreads();
}

#if USE_TDM
__device__ __forceinline__ unsigned int lds_offset_of(const void* p) {
    return (unsigned int)(uintptr_t)(__attribute__((address_space(3))) const char*)p;
}

// 2-D TDM tile load: Din f16 per row, 32 rows, row stride = strideElems f16,
// LDS padding of 4 DWORDs (8 f16) appended after every row (pad_interval code).
__device__ __forceinline__ void tdm_load_tile(const f16* gsrc, unsigned int lds_off,
                                              int din, int padCode, long long strideElems) {
    unsigned long long ga = (unsigned long long)(uintptr_t)gsrc;
    u32x4 g0;
    g0[0] = 1u;                                            // count=1 (valid user D#)
    g0[1] = lds_off;                                       // lds_addr
    g0[2] = (unsigned int)(ga & 0xffffffffu);              // global_addr[31:0]
    g0[3] = (unsigned int)((ga >> 32) & 0x01ffffffu)       // global_addr[56:32]
          | (2u << 30);                                    // type = 2 (image/tensor)
    i32x8 g1;
    g1[0] = (1 << 16)                                      // data_size = 2 bytes
          | (1 << 20)                                      // pad_enable
          | (padCode << 22)                                // pad_interval (one row)
          | (3 << 25);                                     // pad_amount = 4 DWORDs
    g1[1] = (din & 0xffff) << 16;                          // tensor_dim0[15:0]
    g1[2] = ((din >> 16) & 0xffff) | (32 << 16);           // tensor_dim0 hi | tensor_dim1=32
    g1[3] = (din & 0xffff) << 16;                          // tile_dim0 = din
    g1[4] = 32;                                            // tile_dim1 = 32, tile_dim2 = 0
    g1[5] = (int)(strideElems & 0xffffffffll);             // tensor_dim0_stride lo
    g1[6] = (int)((strideElems >> 32) & 0xffffll);         // tensor_dim0_stride hi
    g1[7] = 0;
    i32x4 z4 = {0, 0, 0, 0};
#if __clang_major__ >= 23
    i32x8 z8 = {};
    __builtin_amdgcn_tensor_load_to_lds(g0, g1, z4, z4, z8, 0);
#else
    __builtin_amdgcn_tensor_load_to_lds(g0, g1, z4, z4, 0);
#endif
}
#endif

constexpr size_t lstm_smem(int K) {
    int DIN = K - HID;
    return (size_t)K * 256                 // W fragments (f16)
         + (size_t)2 * 32 * (DIN + 8) * 2  // X double buffer (f16, padded rows)
         + (size_t)32 * (HID + 8) * 2      // H buffer (f16, padded rows)
         + (size_t)32 * GP * 4             // gate buffer (f32)
         + (size_t)32 * 33 * 4             // c state (f32)
         + (size_t)LCOLS * 4;              // bias slice
}

// ---------------------------------------------------------------------------
// Persistent bidirectional LSTM layer.  Grid = 16 WGs x 256 threads:
// blocks 0..7 forward, 8..15 reverse; 8 WGs per direction cooperate per step
// via a device-scope barrier.  Weight slab LDS-resident for all 2048 steps;
// x tiles DMA'd one step ahead by the Tensor Data Mover (double-buffered).
// ---------------------------------------------------------------------------
template <int K>
__global__ void lstm_dirpair(const f16*   __restrict__ xsrc,  // (B,T,DIN) f16
                             const f16*   __restrict__ PW,    // pair base, dir stride K*GATES
                             const float* __restrict__ bias,  // pair base, dir stride GATES
                             const float* __restrict__ h0,    // pair base or null
                             const float* __restrict__ c0,
                             float*       __restrict__ hfin,  // pair base, dir stride 8192
                             float*       __restrict__ cfin,
                             f16*         __restrict__ hbuf,  // pair base, dir stride 2*8192
                             f16*         __restrict__ outp,  // (B,T,512) f16
                             int*         __restrict__ syncc) // pair base, dir stride 1
{
    constexpr int DIN  = K - HID;
    constexpr int XPAD = DIN + 8;
    constexpr int HPAD = HID + 8;
    constexpr int PADC = (DIN == 512) ? 7 : 5;   // row DWORDs: 256 -> 7, 64 -> 5

    extern __shared__ char smem[];
    f16*   Wl = (f16*)smem;                                    // K*128 f16
    f16*   Xl = (f16*)(smem + (size_t)K * 256);                // 2 x 32 x XPAD f16
    f16*   Hl = (f16*)(smem + (size_t)K * 256 + (size_t)2 * 32 * XPAD * 2);
    float* Gl = (float*)((char*)Hl + (size_t)32 * HPAD * 2);
    float* Cl = Gl + 32 * GP;
    float* Bl = Cl + 32 * 33;

    const int tid  = threadIdx.x;
    const int lane = tid & 31;
    const int wave = tid >> 5;      // = col tile ct (0..7)
    const int wg   = blockIdx.x & 7;
    const int dir  = blockIdx.x >> 3;

    const f16*   PWi   = PW   + (size_t)dir * K * GATES + (size_t)wg * K * LCOLS;
    const float* biasi = bias + dir * GATES;
    f16*         hb    = hbuf + dir * 2 * 8192;
    int*         cnt   = syncc + dir;
    f16*         outi  = outp + dir * HID;   // feature offset for this direction

    // ---- one-time init: weights -> LDS, bias slice, c state, h state ----
    {
        const v8h* src = (const v8h*)PWi;
        v8h*       dst = (v8h*)Wl;
        for (int i = tid; i < K * LCOLS / 8; i += 256) dst[i] = src[i];
    }
    for (int lc = tid; lc < LCOLS; lc += 256) {
        int gt = lc >> 5, j = lc & 31;
        Bl[lc] = biasi[gt * HID + wg * HS + j];
    }
    for (int e = tid; e < BATCH * HS; e += 256) {
        int b = e >> 5, hu = e & 31;
        Cl[b * 33 + hu] = c0 ? c0[dir * 8192 + b * HID + wg * HS + hu] : 0.0f;
    }
    if (wg == 0) {
        for (int e = tid; e < BATCH * HID; e += 256)
            hb[e] = (f16)(h0 ? h0[dir * 8192 + e] : 0.0f);
    }
#if USE_TDM
    const unsigned int xoff0 = lds_offset_of(Xl);
    const unsigned int xoff1 = xoff0 + 32 * XPAD * 2;
    if (wave == 0) {   // async-prefetch x(0) into X buffer 0
        int tt0 = dir ? TLEN - 1 : 0;
        tdm_load_tile(xsrc + (size_t)tt0 * DIN, xoff0, DIN, PADC, (long long)TLEN * DIN);
    }
#endif
    grid_barrier(cnt, NWG * 1);

    // ---- sequential recurrence ----
    for (int t = 0; t < TLEN; ++t) {
        const int tt = dir ? (TLEN - 1 - t) : t;
        f16* Xc = Xl + (t & 1) * (32 * XPAD);

        // stage h(t) into LDS (16 KB, from the double-buffered global h)
        {
            const v8h* hsrc = (const v8h*)(hb + (t & 1) * 8192);
            for (int i = tid; i < BATCH * HID / 8; i += 256) {
                int b = i >> 5, dc = i & 31;
                *(v8h*)(Hl + b * HPAD + dc * 8) = hsrc[i];
            }
        }
#if USE_TDM
        // issue TDM for x(t+1) into the other X buffer, then wait so that only
        // that newest DMA is outstanding => x(t) is complete (TDM is in-order).
        if (wave == 0) {
            int tn  = (t + 1 < TLEN) ? t + 1 : t;           // dummy on last step
            int ttn = dir ? (TLEN - 1 - tn) : tn;
            tdm_load_tile(xsrc + (size_t)ttn * DIN, (t & 1) ? xoff0 : xoff1,
                          DIN, PADC, (long long)TLEN * DIN);
            __builtin_amdgcn_s_wait_tensorcnt(1);
        }
#else
        // synchronous fallback staging of x(t)
        for (int i = tid; i < BATCH * DIN / 8; i += 256) {
            int b = i / (DIN / 8), dc = i % (DIN / 8);
            *(v8h*)(Xc + b * XPAD + dc * 8) =
                ((const v8h*)xsrc)[(size_t)(b * TLEN + tt) * (DIN / 8) + dc];
        }
        if (t + 1 < TLEN) {
            int tn = dir ? (TLEN - 2 - t) : (t + 1);
            __builtin_prefetch(xsrc + (size_t)((tid >> 3) * TLEN + tn) * DIN +
                               (size_t)(tid & 7) * (DIN / 8), 0, 1);
        }
#endif
        __syncthreads();

        // WMMA GEMM: (32 x K) x (K x 128) -> two 16x16 row tiles per wave
        v8f acc0 = {}; v8f acc1 = {};
        {
            const int m0  = lane & 15;
            const int kb0 = (lane >> 4) * 8;
            #pragma unroll 4
            for (int kb = 0; kb < DIN / 32; ++kb) {
                const v16h bf = *(const v16h*)(Wl + ((((kb * 8 + wave) * 32) + lane) << 4));
                const f16* a0 = Xc + m0 * XPAD + kb * 32 + kb0;
                const f16* a1 = a0 + 16 * XPAD;
                v8h a0lo = *(const v8h*)a0;        v8h a0hi = *(const v8h*)(a0 + 16);
                v8h a1lo = *(const v8h*)a1;        v8h a1hi = *(const v8h*)(a1 + 16);
                v16h af0 = __builtin_shufflevector(a0lo, a0hi, 0,1,2,3,4,5,6,7,8,9,10,11,12,13,14,15);
                v16h af1 = __builtin_shufflevector(a1lo, a1hi, 0,1,2,3,4,5,6,7,8,9,10,11,12,13,14,15);
                acc0 = __builtin_amdgcn_wmma_f32_16x16x32_f16(false, af0, false, bf, (short)0, acc0, false, false);
                acc1 = __builtin_amdgcn_wmma_f32_16x16x32_f16(false, af1, false, bf, (short)0, acc1, false, false);
            }
            #pragma unroll 4
            for (int kb = 0; kb < HID / 32; ++kb) {
                const v16h bf = *(const v16h*)(Wl + (((((DIN / 32 + kb) * 8 + wave) * 32) + lane) << 4));
                const f16* a0 = Hl + m0 * HPAD + kb * 32 + kb0;
                const f16* a1 = a0 + 16 * HPAD;
                v8h a0lo = *(const v8h*)a0;        v8h a0hi = *(const v8h*)(a0 + 16);
                v8h a1lo = *(const v8h*)a1;        v8h a1hi = *(const v8h*)(a1 + 16);
                v16h af0 = __builtin_shufflevector(a0lo, a0hi, 0,1,2,3,4,5,6,7,8,9,10,11,12,13,14,15);
                v16h af1 = __builtin_shufflevector(a1lo, a1hi, 0,1,2,3,4,5,6,7,8,9,10,11,12,13,14,15);
                acc0 = __builtin_amdgcn_wmma_f32_16x16x32_f16(false, af0, false, bf, (short)0, acc0, false, false);
                acc1 = __builtin_amdgcn_wmma_f32_16x16x32_f16(false, af1, false, bf, (short)0, acc1, false, false);
            }
        }
        // scatter accumulators to LDS gate buffer (C layout: VGPR j -> M=j+8*(lane>=16))
        {
            int n  = wave * 16 + (lane & 15);
            int mb = (lane >> 4) * 8;
            #pragma unroll
            for (int j = 0; j < 8; ++j) {
                Gl[(mb + j) * GP + n]      = acc0[j];
                Gl[(16 + mb + j) * GP + n] = acc1[j];
            }
        }
        __syncthreads();

        // elementwise LSTM cell update for this WG's 32 hidden units
        #pragma unroll
        for (int r = 0; r < 4; ++r) {
            int e = r * 256 + tid;
            int b = e >> 5, hu = e & 31;
            float gi = Gl[b * GP + hu]       + Bl[hu];
            float gf = Gl[b * GP + 32 + hu]  + Bl[32 + hu];
            float gg = Gl[b * GP + 64 + hu]  + Bl[64 + hu];
            float go = Gl[b * GP + 96 + hu]  + Bl[96 + hu];
            float c  = Cl[b * 33 + hu];
            float si = 1.0f / (1.0f + __expf(-gi));
            float sf = 1.0f / (1.0f + __expf(-gf));
            float so = 1.0f / (1.0f + __expf(-go));
            float cn = sf * c + si * tanhf(gg);
            float hv = so * tanhf(cn);
            Cl[b * 33 + hu] = cn;
            hb[((t + 1) & 1) * 8192 + b * HID + wg * HS + hu] = (f16)hv;
            outi[(size_t)(b * TLEN + tt) * 512 + wg * HS + hu] = (f16)hv;
            if (t == TLEN - 1) {
                hfin[dir * 8192 + b * HID + wg * HS + hu] = hv;
                cfin[dir * 8192 + b * HID + wg * HS + hu] = cn;
            }
        }
        grid_barrier(cnt, NWG * (t + 2));
    }
}

// y[b,t] = tanh(out2[b,t,:] . fc_w + fc_b)
__global__ void fc_kernel(const f16* __restrict__ out2,
                          const float* __restrict__ fcw,
                          const float* __restrict__ fcb,
                          float* __restrict__ y) {
    __shared__ float wl[512];
    int tid = threadIdx.x;
    for (int i = tid; i < 512; i += 256) wl[i] = fcw[i];
    __syncthreads();
    int id = blockIdx.x * 256 + tid;
    int b = id >> 11, t = id & (TLEN - 1);
    const v8h* row = (const v8h*)(out2 + (size_t)(b * TLEN + t) * 512);
    float acc = 0.0f;
    #pragma unroll 8
    for (int j = 0; j < 64; ++j) {
        v8h v = row[j];
        #pragma unroll
        for (int e = 0; e < 8; ++e) acc += (float)v[e] * wl[j * 8 + e];
    }
    y[id] = tanhf(acc + fcb[0]);
}

extern "C" void kernel_launch(void* const* d_in, const int* in_sizes, int n_in,
                              void* d_out, int out_size, void* d_ws, size_t ws_size,
                              hipStream_t stream) {
    (void)in_sizes; (void)n_in; (void)out_size; (void)ws_size;
    const float* x     = (const float*)d_in[0];
    const float* wih1f = (const float*)d_in[1];
    const float* whh1f = (const float*)d_in[2];
    const float* bih1f = (const float*)d_in[3];
    const float* bhh1f = (const float*)d_in[4];
    const float* wih1r = (const float*)d_in[5];
    const float* whh1r = (const float*)d_in[6];
    const float* bih1r = (const float*)d_in[7];
    const float* bhh1r = (const float*)d_in[8];
    const float* wih2f = (const float*)d_in[9];
    const float* whh2f = (const float*)d_in[10];
    const float* bih2f = (const float*)d_in[11];
    const float* bhh2f = (const float*)d_in[12];
    const float* wih2r = (const float*)d_in[13];
    const float* whh2r = (const float*)d_in[14];
    const float* bih2r = (const float*)d_in[15];
    const float* bhh2r = (const float*)d_in[16];
    const float* fcw   = (const float*)d_in[17];
    const float* fcb   = (const float*)d_in[18];

    // ---- workspace carve ----
    size_t off = 0;
    auto carve = [&](size_t bytes) {
        off = (off + 255) & ~(size_t)255;
        void* p = (char*)d_ws + off;
        off += bytes;
        return p;
    };
    int*   syncc = (int*)  carve(256);                                   // 4 counters
    float* biasW = (float*)carve((size_t)4 * GATES * sizeof(float));
    f16*   PW    = (f16*)  carve((size_t)(2 * 384 + 2 * 768) * GATES * sizeof(f16));
    f16*   hbuf  = (f16*)  carve((size_t)4 * 2 * 8192 * sizeof(f16));
    float* hfin  = (float*)carve((size_t)4 * 8192 * sizeof(float));
    float* cfin  = (float*)carve((size_t)4 * 8192 * sizeof(float));
    f16*   xh    = (f16*)  carve((size_t)BATCH * TLEN * 128 * sizeof(f16));
    f16*   out1  = (f16*)  carve((size_t)BATCH * TLEN * 512 * sizeof(f16));
    f16*   out2  = (f16*)  carve((size_t)BATCH * TLEN * 512 * sizeof(f16));

    hipMemsetAsync(syncc, 0, 256, stream);   // barrier counters must start at 0

    // x -> f16 (TDM moves raw bytes; convert once up front)
    cvt_f32_to_f16<<<(BATCH * TLEN * 128 / 4 + 255) / 256, 256, 0, stream>>>(
        x, xh, BATCH * TLEN * 128 / 4);

    // weight packing + bias sums (tiny, parallel)
    const int K1 = 384, K2 = 768;
    pack_weights<<<(K1 * GATES + 255) / 256, 256, 0, stream>>>(wih1f, whh1f, 128, K1, PW);
    pack_weights<<<(K1 * GATES + 255) / 256, 256, 0, stream>>>(wih1r, whh1r, 128, K1, PW + (size_t)K1 * GATES);
    pack_weights<<<(K2 * GATES + 255) / 256, 256, 0, stream>>>(wih2f, whh2f, 512, K2, PW + (size_t)2 * K1 * GATES);
    pack_weights<<<(K2 * GATES + 255) / 256, 256, 0, stream>>>(wih2r, whh2r, 512, K2, PW + (size_t)2 * K1 * GATES + (size_t)K2 * GATES);
    bias_sum<<<4, 256, 0, stream>>>(bih1f, bhh1f, biasW);
    bias_sum<<<4, 256, 0, stream>>>(bih1r, bhh1r, biasW + GATES);
    bias_sum<<<4, 256, 0, stream>>>(bih2f, bhh2f, biasW + 2 * GATES);
    bias_sum<<<4, 256, 0, stream>>>(bih2r, bhh2r, biasW + 3 * GATES);

    const size_t smem1 = lstm_smem(K1);   // ~154 KB  (<= 320 KB WGP LDS)
    const size_t smem2 = lstm_smem(K2);   // ~302 KB
    hipFuncSetAttribute(reinterpret_cast<const void*>(&lstm_dirpair<384>),
                        hipFuncAttributeMaxDynamicSharedMemorySize, (int)smem1);
    hipFuncSetAttribute(reinterpret_cast<const void*>(&lstm_dirpair<768>),
                        hipFuncAttributeMaxDynamicSharedMemorySize, (int)smem2);

    // layer 1 (fwd+rev concurrently), zero initial state
    lstm_dirpair<384><<<dim3(16), dim3(256), smem1, stream>>>(
        xh, PW, biasW, nullptr, nullptr, hfin, cfin, hbuf, out1, syncc);

    // layer 2, initial state = layer-1 final state per direction
    lstm_dirpair<768><<<dim3(16), dim3(256), smem2, stream>>>(
        out1, PW + (size_t)2 * K1 * GATES, biasW + 2 * GATES,
        hfin, cfin, hfin + 2 * 8192, cfin + 2 * 8192,
        hbuf + 2 * 2 * 8192, out2, syncc + 2);

    // FC head + tanh -> (B,1,T) f32
    fc_kernel<<<(BATCH * TLEN) / 256, 256, 0, stream>>>(out2, fcw, fcb, (float*)d_out);
}
